// SpatiotemporalGNN_14328010899636
// MI455X (gfx1250) — matrix operand order repaired and checked by reading, not compiled, not run on hardware
//
#include <hip/hip_runtime.h>
#include <hip/hip_bf16.h>

// Spatiotemporal GNN for MI455X (gfx1250).
// Dominant work: two A[4096x4096] @ X[4096x64] passes per batch (B=8),
// mapped onto v_wmma_f32_16x16x32_bf16 (fp32 accumulate). A converted to
// bf16 once; 32MB resident in the 192MB L2 for both passes.
// A-tiles staged with GLOBAL_LOAD_ASYNC_TO_LDS_B128 (ASYNCcnt) into a
// TRIPLE-buffered LDS pipeline (lookahead 2): while tile T is computed,
// tile T+1 is resident and tile T+2 is in flight; in-order ASYNCcnt lets
// s_wait_asynccnt 2 release the barrier without draining the pipe.

typedef __attribute__((ext_vector_type(16))) __bf16 v16bf;
typedef __attribute__((ext_vector_type(8)))  float v8f;

union FragU { uint4 q[2]; v16bf v; };

static __device__ __forceinline__ unsigned short f2bf_bits(float f) {
  union { float f; unsigned u; } v; v.f = f;
  unsigned x = v.u;
  x += 0x7FFFu + ((x >> 16) & 1u);   // round to nearest even
  return (unsigned short)(x >> 16);
}
static __device__ __forceinline__ unsigned pack2(float lo, float hi) {
  return ((unsigned)f2bf_bits(hi) << 16) | (unsigned)f2bf_bits(lo);
}

// ---------------------------------------------------------------------------
// A (fp32) -> bf16, 8 elements per thread
__global__ __launch_bounds__(256) void k_convert_bf16(
    const float* __restrict__ src, unsigned short* __restrict__ dst) {
  size_t i = ((size_t)blockIdx.x * 256 + threadIdx.x) * 8;
  float4 a = *(const float4*)(src + i);
  float4 b = *(const float4*)(src + i + 4);
  uint4 o;
  o.x = pack2(a.x, a.y); o.y = pack2(a.z, a.w);
  o.z = pack2(b.x, b.y); o.w = pack2(b.z, b.w);
  *(uint4*)(dst + i) = o;
}

// ---------------------------------------------------------------------------
// xw1 = x @ W1  (fp32 math, bf16 output).  x: [32768,40], W1: [40,64]
__global__ __launch_bounds__(256) void k_xw1(
    const float* __restrict__ x, const float* __restrict__ W1,
    unsigned short* __restrict__ out) {
  __shared__ float w[40 * 64];
  __shared__ float xs[4 * 40];
  const int tid = threadIdx.x;
  const int row0 = blockIdx.x * 4;
  for (int i = tid; i < 40 * 64; i += 256) w[i] = W1[i];
  for (int i = tid; i < 4 * 40; i += 256) xs[i] = x[(size_t)row0 * 40 + i];
  __syncthreads();
  const int r = tid >> 6, c = tid & 63;
  float acc = 0.f;
#pragma unroll
  for (int k = 0; k < 40; ++k) acc += xs[r * 40 + k] * w[k * 64 + c];
  out[(size_t)(row0 + r) * 64 + c] = f2bf_bits(acc);
}

// ---------------------------------------------------------------------------
// hw2 = h @ W2  (fp32 math, bf16 output).  h: [32768,64], W2: [64,64]
__global__ __launch_bounds__(256) void k_h_w2(
    const float* __restrict__ h, const float* __restrict__ W2,
    unsigned short* __restrict__ out) {
  __shared__ float w[64 * 64];
  __shared__ float xs[4 * 64];
  const int tid = threadIdx.x;
  const int row0 = blockIdx.x * 4;
  for (int i = tid; i < 64 * 64; i += 256) w[i] = W2[i];
  for (int i = tid; i < 4 * 64; i += 256) xs[i] = h[(size_t)row0 * 64 + i];
  __syncthreads();
  const int r = tid >> 6, c = tid & 63;
  float acc = 0.f;
#pragma unroll
  for (int k = 0; k < 64; ++k) acc += xs[r * 64 + k] * w[k * 64 + c];
  out[(size_t)(row0 + r) * 64 + c] = f2bf_bits(acc);
}

// ---------------------------------------------------------------------------
// C[b] = A_bf16[4096x4096] @ X_bf16[b][4096x64] + bias   (fp32 out)
// Block: 256 threads (8 waves). Block tile: 128 rows x 64 cols.
// Wave w computes rows [w*16, w*16+16) x all 64 cols: 4 WMMA accumulators.
// Triple-buffered LDS; A tiles staged by async-to-LDS, lookahead 2.
#define LDA 40              // 32 + 8 bf16 pad -> conflict-free b128 frag reads
#define ABH (128 * LDA)     // A buffer stage (elements);  bytes = 10240
#define XBH (64 * LDA)      // X buffer stage (elements);  bytes = 5120
#define ABYT (ABH * 2)
#define NT 128              // K tiles (4096 / 32)
__global__ __launch_bounds__(256) void k_gemm_bf16(
    const unsigned short* __restrict__ Abf,
    const unsigned short* __restrict__ Xbf,
    const float* __restrict__ bias,
    float* __restrict__ C) {
  __shared__ unsigned short As[3 * ABH];  // A tiles, row-major (padded)
  __shared__ unsigned short Xs[3 * XBH];  // X tiles, TRANSPOSED: Xs[col][k]

  const int tid  = threadIdx.x;
  const int lane = tid & 31;
  const int wave = tid >> 5;
  const int m0   = blockIdx.x * 128;
  const unsigned short* Xb = Xbf + (size_t)blockIdx.y * (4096 * 64);
  float*                Cb = C   + (size_t)blockIdx.y * (4096 * 64);

  v8f acc[4];
#pragma unroll
  for (int t = 0; t < 4; ++t)
#pragma unroll
    for (int i = 0; i < 8; ++i) acc[t][i] = 0.0f;

  // --- A tile async staging: thread owns row ar, 16 elems at ac (2 x b128)
  const int ar = tid >> 1;
  const int ac = (tid & 1) * 16;
  unsigned agoff = (unsigned)(((m0 + ar) * 4096 + ac) * 2);      // global bytes
  const unsigned alds0 = (unsigned)(size_t)(As + ar * LDA + ac); // LDS bytes (buf0)

  // --- X tile staging through VGPRs (transpose): 4 cols x 2 k-rows / thread
  const int xc4 = (tid & 15) * 4;   // column base
  const int xkr = (tid >> 4) * 2;   // even k-row
  const unsigned short* xg = Xb + (size_t)xkr * 64 + xc4;

  // --- fragment addressing (ISA 16-bit A / B layouts, wave32)
  const int mloc = wave * 16 + (lane & 15);
  const int c0a  = (lane >> 4) * 8;    // A frag: K chunks [c0,c0+8),[c0+16,c0+24)
  const int kb   = (lane >> 4) * 8;    // B frag: K chunks [kb,kb+8),[kb+16,kb+24)

  // ---- prologue: A tiles 0,1 in flight; X tile 0 staged, X tile 1 in regs
  asm volatile(
      "global_load_async_to_lds_b128 %0, %1, %2\n\t"
      "global_load_async_to_lds_b128 %0, %1, %2 offset:16"
      :: "v"(alds0), "v"(agoff), "s"(Abf) : "memory");
  asm volatile(
      "global_load_async_to_lds_b128 %0, %1, %2\n\t"
      "global_load_async_to_lds_b128 %0, %1, %2 offset:16"
      :: "v"(alds0 + (unsigned)ABYT), "v"(agoff + 64u), "s"(Abf) : "memory");
  agoff += 128u;                     // next issue = tile 2
  {
    uint2 r0 = *(const uint2*)xg;
    uint2 r1 = *(const uint2*)(xg + 64);
    const unsigned short* e0 = (const unsigned short*)&r0;
    const unsigned short* e1 = (const unsigned short*)&r1;
#pragma unroll
    for (int j = 0; j < 4; ++j)
      *(unsigned*)(Xs + (xc4 + j) * LDA + xkr) =
          (unsigned)e0[j] | ((unsigned)e1[j] << 16);
  }
  uint2 n0 = *(const uint2*)(xg + 32 * 64);        // X tile 1 held in regs
  uint2 n1 = *(const uint2*)(xg + 32 * 64 + 64);
  const unsigned short* xgn = xg + 2 * 32 * 64;    // next X load = tile 2
  asm volatile("s_wait_asynccnt 2" ::: "memory");  // tile 0 resident
  __syncthreads();

  int idx0 = 0, idx1 = 1, idx2 = 2;  // compute / X-store / A-async buffers
  for (int T = 0; T < NT; ++T) {
    const bool have2 = (T + 2) < NT;   // block-uniform
    uint2 t0 = {}, t1 = {};
    if (have2) {
      // issue A tile T+2 async copy + X tile T+2 global loads NOW
      const unsigned nlds = alds0 + (unsigned)(idx2 * ABYT);
      asm volatile(
          "global_load_async_to_lds_b128 %0, %1, %2\n\t"
          "global_load_async_to_lds_b128 %0, %1, %2 offset:16"
          :: "v"(nlds), "v"(agoff), "s"(Abf) : "memory");
      agoff += 64u;
      t0 = *(const uint2*)xgn;
      t1 = *(const uint2*)(xgn + 64);
      xgn += 32 * 64;
    }

    // ---- compute tile T on buffer idx0: group all frag loads, then 4 WMMAs
    const unsigned short* Ac = As + idx0 * ABH;
    const unsigned short* Xc = Xs + idx0 * XBH;
    FragU fa, fb[4];
    fa.q[0] = *(const uint4*)(Ac + mloc * LDA + c0a);
    fa.q[1] = *(const uint4*)(Ac + mloc * LDA + c0a + 16);
#pragma unroll
    for (int t = 0; t < 4; ++t) {
      const int col = t * 16 + (lane & 15);
      fb[t].q[0] = *(const uint4*)(Xc + col * LDA + kb);
      fb[t].q[1] = *(const uint4*)(Xc + col * LDA + kb + 16);
    }
#pragma unroll
    for (int t = 0; t < 4; ++t)
      acc[t] = __builtin_amdgcn_wmma_f32_16x16x32_bf16(
          false, fa.v, false, fb[t].v, (short)0, acc[t], false, false);

    // ---- transpose-store X tile T+1 (regs loaded last iteration)
    if (T + 1 < NT) {
      unsigned short* Xd = Xs + idx1 * XBH;
      const unsigned short* e0 = (const unsigned short*)&n0;
      const unsigned short* e1 = (const unsigned short*)&n1;
#pragma unroll
      for (int j = 0; j < 4; ++j)
        *(unsigned*)(Xd + (xc4 + j) * LDA + xkr) =
            (unsigned)e0[j] | ((unsigned)e1[j] << 16);
    }
    n0 = t0; n1 = t1;

    // tile T+1's A copies complete (in-order ASYNCcnt); T+2 stays in flight
    if (have2) asm volatile("s_wait_asynccnt 2" ::: "memory");
    else       asm volatile("s_wait_asynccnt 0" ::: "memory");
    __syncthreads();
    idx0 = (idx0 == 2) ? 0 : idx0 + 1;
    idx1 = (idx1 == 2) ? 0 : idx1 + 1;
    idx2 = (idx2 == 2) ? 0 : idx2 + 1;
  }

  // D layout: lanes 0-15 -> M=r, lanes 16-31 -> M=r+8 ; N = lane&15 (+tile)
  const int mst = wave * 16 + ((lane >> 4) ? 8 : 0);
#pragma unroll
  for (int t = 0; t < 4; ++t) {
    const int n = t * 16 + (lane & 15);
    const float bv = bias[n];
#pragma unroll
    for (int r = 0; r < 8; ++r)
      Cb[(size_t)(m0 + mst + r) * 64 + n] = acc[t][r] + bv;
  }
}

// ---------------------------------------------------------------------------
// Column sums of h2[b] over all 4096 rows -> colsum[b][64]
__global__ __launch_bounds__(256) void k_colsum(
    const float* __restrict__ H2, float* __restrict__ colsum) {
  __shared__ float red[256];
  const int b = blockIdx.x, tid = threadIdx.x;
  const float* Hb = H2 + (size_t)b * 4096 * 64;
  const int c = tid & 63, r0 = tid >> 6;
  float acc = 0.f;
  for (int n = r0; n < 4096; n += 4) acc += Hb[(size_t)n * 64 + c];
  red[tid] = acc;
  __syncthreads();
  if (tid < 64)
    colsum[b * 64 + tid] = red[tid] + red[tid + 64] + red[tid + 128] + red[tid + 192];
}

// ---------------------------------------------------------------------------
// mean_t(conv1d) collapsed: S[i,k] = shifted column means; then conv_w/fc.
__global__ __launch_bounds__(64) void k_head(
    const float* __restrict__ H2, const float* __restrict__ colsum,
    const float* __restrict__ conv_w, const float* __restrict__ conv_b,
    const float* __restrict__ fc_w, const float* __restrict__ fc_b,
    float* __restrict__ out) {
  __shared__ float S[64 * 5];
  __shared__ float red[64];
  const int b = blockIdx.x, tid = threadIdx.x;   // 64 threads
  const float* Hb = H2 + (size_t)b * 4096 * 64;

  // phase 1: thread = channel i
  float head[4], tail[4];
#pragma unroll
  for (int j = 0; j < 4; ++j) head[j] = Hb[(size_t)j * 64 + tid];
#pragma unroll
  for (int j = 0; j < 4; ++j) tail[j] = Hb[(size_t)(4092 + j) * 64 + tid];
  const float total = colsum[b * 64 + tid];
  const float inv = 1.0f / 4092.0f;
  float pre = 0.f;
#pragma unroll
  for (int k = 0; k < 5; ++k) {
    float post = 0.f;
    for (int j = k; j < 4; ++j) post += tail[j];   // rows >= k+4092
    S[tid * 5 + k] = (total - pre - post) * inv;
    if (k < 4) pre += head[k];                     // rows < k
  }
  __syncthreads();

  // phase 2: thread = output channel o
  float acc = conv_b[tid];
  for (int i = 0; i < 64; ++i)
#pragma unroll
    for (int k = 0; k < 5; ++k)
      acc += conv_w[tid * 320 + i * 5 + k] * S[i * 5 + k];
  red[tid] = acc * fc_w[tid];   // F_OUT == 1
  __syncthreads();
  for (int s = 32; s > 0; s >>= 1) {
    if (tid < s) red[tid] += red[tid + s];
    __syncthreads();
  }
  if (tid == 0) out[b] = red[0] + fc_b[0];
}

// ---------------------------------------------------------------------------
extern "C" void kernel_launch(void* const* d_in, const int* in_sizes, int n_in,
                              void* d_out, int out_size, void* d_ws, size_t ws_size,
                              hipStream_t stream) {
  const float* x      = (const float*)d_in[0];
  const float* A      = (const float*)d_in[1];
  const float* W1     = (const float*)d_in[2];
  const float* b1     = (const float*)d_in[3];
  const float* W2     = (const float*)d_in[4];
  const float* b2     = (const float*)d_in[5];
  const float* conv_w = (const float*)d_in[6];
  const float* conv_b = (const float*)d_in[7];
  const float* fc_w   = (const float*)d_in[8];
  const float* fc_b   = (const float*)d_in[9];
  float* out = (float*)d_out;

  char* ws = (char*)d_ws;
  unsigned short* A_bf   = (unsigned short*)(ws);               // 32 MB
  unsigned short* act_bf = (unsigned short*)(ws + 33554432);    //  4 MB
  float*          act_f  = (float*)(ws + 37748736);             //  8 MB
  float*          colsum = (float*)(ws + 46137344);             //  2 KB

  // 1) A -> bf16 (16.78M elems, 8/thread)
  k_convert_bf16<<<8192, 256, 0, stream>>>(A, A_bf);
  // 2) xw1 = x @ W1 -> bf16   (32768 rows, 4 rows/block)
  k_xw1<<<8192, 256, 0, stream>>>(x, W1, act_bf);
  // 3) h1 = A @ xw1 + b1 -> fp32  (grid: 32 M-blocks x 8 batches)
  k_gemm_bf16<<<dim3(32, 8), 256, 0, stream>>>(A_bf, act_bf, b1, act_f);
  // 4) hw2 = h1 @ W2 -> bf16
  k_h_w2<<<8192, 256, 0, stream>>>(act_f, W2, act_bf);
  // 5) h2 = A @ hw2 + b2 -> fp32
  k_gemm_bf16<<<dim3(32, 8), 256, 0, stream>>>(A_bf, act_bf, b2, act_f);
  // 6) column sums of h2
  k_colsum<<<8, 256, 0, stream>>>(act_f, colsum);
  // 7) collapsed conv1d-mean + FC head
  k_head<<<8, 64, 0, stream>>>(act_f, colsum, conv_w, conv_b, fc_w, fc_b, out);
}
